// DEPTHVIT_35124242546904
// MI455X (gfx1250) — compile-verified, matrix operand order checked by reading, not run on hardware
//
#include <hip/hip_runtime.h>
#include <hip/hip_bf16.h>

typedef __attribute__((ext_vector_type(16))) _Float16 v16h;
typedef __attribute__((ext_vector_type(8)))  float    v8f;

#define C_DIM   768
#define N_TOK   480
#define SEQ     481
#define B_SZ    128
#define Q_SCALE 0.05103f
#define LN_EPS  1e-6f
#define INV_HALF_CHUNKS (1.0f / 48.0f)   // NUM_CHUNKS/2 = 48

// ---------------------------------------------------------------------------
// Async global -> LDS helpers (CDNA5, ASYNCcnt-tracked).
// One call moves a 3072-byte token row: 32 lanes x 16 B x 6 instructions.
// INST_OFFSET is added to BOTH the LDS and global addresses (ISA 10. async),
// so a single (lds, gaddr) pair with stepped offsets covers the row.
// ---------------------------------------------------------------------------
__device__ __forceinline__ void dv_async_row(const float* __restrict__ g,
                                             unsigned lds_off, int lane) {
  const char* gp = (const char*)g + lane * 16;
  unsigned    lo = lds_off + lane * 16;
  asm volatile(
      "global_load_async_to_lds_b128 %0, %1, off offset:0\n\t"
      "global_load_async_to_lds_b128 %0, %1, off offset:512\n\t"
      "global_load_async_to_lds_b128 %0, %1, off offset:1024\n\t"
      "global_load_async_to_lds_b128 %0, %1, off offset:1536\n\t"
      "global_load_async_to_lds_b128 %0, %1, off offset:2048\n\t"
      "global_load_async_to_lds_b128 %0, %1, off offset:2560"
      :: "v"(lo), "v"(gp) : "memory");
}
__device__ __forceinline__ void dv_wait_async_le6() {
  asm volatile("s_wait_asynccnt 0x6" ::: "memory");
}
__device__ __forceinline__ void dv_wait_async_0() {
  asm volatile("s_wait_asynccnt 0x0" ::: "memory");
}

// ---------------------------------------------------------------------------
// WMMA GEMM:  D[M=128, N=768] = alpha * A[128 x 768] * B + bias
//   TRANSB=1: B[k,n] = Bp[n*ldb + k]; TRANSB=0: B[k,n] = Bp[k*ldb + n]
// One wave per 16x16 tile, f32->f16 inline, f32 accumulate, K unrolled x4.
// A frag (ISA 7.12.2): lane%16 = M, elem e -> K = e + (e>=8?8:0) + (lane>=16?8:0).
// B frag mirrored (lane%16 = N). C/D: VGPR r -> M = r + (lane>=16?8:0), N = lane%16.
// ---------------------------------------------------------------------------
template <int TRANSB>
__global__ __launch_bounds__(32)
void dv_wmma_gemm(const float* __restrict__ A, int lda,
                  const float* __restrict__ Bp, int ldb,
                  float alpha, const float* __restrict__ bias,
                  float* __restrict__ D, int ldd) {
  const int lane = threadIdx.x;
  const int l16  = lane & 15;
  const int hi8  = (lane >= 16) ? 8 : 0;
  const int n0   = blockIdx.x * 16;
  const int m0   = blockIdx.y * 16;

  const float* arow = A + (m0 + l16) * lda;

  v8f acc = {};
#pragma unroll 4
  for (int k0 = 0; k0 < C_DIM; k0 += 32) {
    v16h af, bf;
#pragma unroll
    for (int e = 0; e < 16; ++e) {
      const int kk = k0 + e + ((e >= 8) ? 8 : 0) + hi8;
      af[e] = (_Float16)arow[kk];
      const float bv = TRANSB ? Bp[(n0 + l16) * ldb + kk]
                              : Bp[kk * ldb + (n0 + l16)];
      bf[e] = (_Float16)bv;
    }
    acc = __builtin_amdgcn_wmma_f32_16x16x32_f16(
        false, af, false, bf, (short)0, acc, false, false);
  }

  const int n = n0 + l16;
  const float bb = bias ? bias[n] : 0.0f;
#pragma unroll
  for (int r = 0; r < 8; ++r) {
    const int m = m0 + r + hi8;
    D[m * ldd + n] = acc[r] * alpha + bb;
  }
}

// ---------------------------------------------------------------------------
// Per-batch streaming kernel. grid = 128, block = 256 (8 wave32s).
//  pass 1: attn[n] = qt[b].x[b,n] — async double-buffered LDS rows per wave
//  chunk : softmax(5) + median-of-5 mask via perm, emit attn/masked outputs
//  pass 2: xw[b] = sum_n w[n]*x[b,n], skipping zero-weight tokens (~L2 hits)
// ---------------------------------------------------------------------------
__global__ __launch_bounds__(256)
void dv_attn_stream(const float* __restrict__ input,
                    const int*   __restrict__ perm,
                    const float* __restrict__ qt,
                    float* __restrict__ xw,
                    float* __restrict__ attn_out,
                    float* __restrict__ masked_out) {
  __shared__ float xrow_s[8][2][C_DIM];   // 48 KB: per-wave double buffer
  __shared__ float qt_s[C_DIM];
  __shared__ float attn_s[N_TOK];
  __shared__ float w_s[N_TOK];
  __shared__ int   perm_s[N_TOK];

  const int b    = blockIdx.x;
  const int tid  = threadIdx.x;
  const int wave = tid >> 5;
  const int lane = tid & 31;
  const float* xb = input + b * (SEQ * C_DIM) + C_DIM;  // tokens 1..480

  for (int i = tid; i < C_DIM; i += 256) qt_s[i] = qt[b * C_DIM + i];
  for (int i = tid; i < N_TOK; i += 256) perm_s[i] = perm[i];
  __syncthreads();

  // ---- pass 1: async double-buffered dot products ----
  const float* r0 = &xrow_s[wave][0][0];
  const float* r1 = &xrow_s[wave][1][0];
  const unsigned o0 = (unsigned)(unsigned long long)r0;  // LDS byte offset
  const unsigned o1 = (unsigned)(unsigned long long)r1;

  dv_async_row(xb + (long)wave * C_DIM, o0, lane);       // preload first row
  int cur = 0;
  for (int n = wave; n < N_TOK; n += 8) {
    const int nn = n + 8;
    if (nn < N_TOK) {
      dv_async_row(xb + (long)nn * C_DIM, cur ? o0 : o1, lane);
      dv_wait_async_le6();                               // older 6 loads done
    } else {
      dv_wait_async_0();
    }
    const float* row = cur ? r1 : r0;
    float acc = 0.0f;
#pragma unroll
    for (int c = lane; c < C_DIM; c += 32) acc += qt_s[c] * row[c];
#pragma unroll
    for (int off = 16; off; off >>= 1) acc += __shfl_xor(acc, off, 32);
    if (lane == 0) attn_s[n] = acc;
    cur ^= 1;
  }
  __syncthreads();

  // ---- chunk softmax + median mask (96 chunks of 5) ----
  if (tid < 96) {
    float p[5]; int idx[5];
    float mx = -1e30f;
#pragma unroll
    for (int j = 0; j < 5; ++j) {
      idx[j] = perm_s[tid * 5 + j];
      p[j]   = attn_s[idx[j]];
      mx     = fmaxf(mx, p[j]);
    }
    float s = 0.0f;
#pragma unroll
    for (int j = 0; j < 5; ++j) { p[j] = __expf(p[j] - mx); s += p[j]; }
    const float inv = 1.0f / s;
#pragma unroll
    for (int j = 0; j < 5; ++j) p[j] *= inv;
    float t[5];
#pragma unroll
    for (int j = 0; j < 5; ++j) t[j] = p[j];
#pragma unroll
    for (int i = 0; i < 4; ++i)
#pragma unroll
      for (int j = 0; j < 4 - i; ++j) {
        const float lo = fminf(t[j], t[j + 1]);
        const float hm = fmaxf(t[j], t[j + 1]);
        t[j] = lo; t[j + 1] = hm;
      }
    const float med = t[2];                               // 3rd smallest of 5
#pragma unroll
    for (int j = 0; j < 5; ++j) {
      attn_out[b * N_TOK + idx[j]] = p[j];
      const float w = (p[j] < med) ? 0.0f : p[j] * INV_HALF_CHUNKS;
      masked_out[b * N_TOK + idx[j]] = w;
      w_s[idx[j]] = w;
    }
  }
  __syncthreads();

  // ---- pass 2: weighted token sum (each thread owns 3 channels) ----
  float a0 = 0.0f, a1 = 0.0f, a2 = 0.0f;
  for (int n = 0; n < N_TOK; ++n) {
    const float w = w_s[n];
    if (w != 0.0f) {
      const float* xr = xb + (long)n * C_DIM;
      __builtin_prefetch(xr + C_DIM, 0, 0);
      a0 += w * xr[tid];
      a1 += w * xr[tid + 256];
      a2 += w * xr[tid + 512];
    }
  }
  xw[b * C_DIM + tid]       = a0;
  xw[b * C_DIM + tid + 256] = a1;
  xw[b * C_DIM + tid + 512] = a2;
}

// ---------------------------------------------------------------------------
// LayerNorm over C=768, one block (8 waves) per batch row.
// ---------------------------------------------------------------------------
__global__ __launch_bounds__(256)
void dv_layernorm(const float* __restrict__ x,
                  const float* __restrict__ gamma,
                  const float* __restrict__ beta,
                  float* __restrict__ out) {
  __shared__ float red[8];
  const int b = blockIdx.x, tid = threadIdx.x;
  const int wave = tid >> 5, lane = tid & 31;
  const float* xr = x + b * C_DIM;
  const float v0 = xr[tid], v1 = xr[tid + 256], v2 = xr[tid + 512];

  float s = v0 + v1 + v2;
#pragma unroll
  for (int off = 16; off; off >>= 1) s += __shfl_xor(s, off, 32);
  if (lane == 0) red[wave] = s;
  __syncthreads();
  float mu = 0.0f;
#pragma unroll
  for (int i = 0; i < 8; ++i) mu += red[i];
  mu *= (1.0f / (float)C_DIM);
  __syncthreads();

  const float d0 = v0 - mu, d1 = v1 - mu, d2 = v2 - mu;
  float sq = d0 * d0 + d1 * d1 + d2 * d2;
#pragma unroll
  for (int off = 16; off; off >>= 1) sq += __shfl_xor(sq, off, 32);
  if (lane == 0) red[wave] = sq;
  __syncthreads();
  float var = 0.0f;
#pragma unroll
  for (int i = 0; i < 8; ++i) var += red[i];
  var *= (1.0f / (float)C_DIM);
  const float rstd = rsqrtf(var + LN_EPS);

  out[b * C_DIM + tid]       = d0 * rstd * gamma[tid]       + beta[tid];
  out[b * C_DIM + tid + 256] = d1 * rstd * gamma[tid + 256] + beta[tid + 256];
  out[b * C_DIM + tid + 512] = d2 * rstd * gamma[tid + 512] + beta[tid + 512];
}

extern "C" void kernel_launch(void* const* d_in, const int* in_sizes, int n_in,
                              void* d_out, int out_size, void* d_ws, size_t ws_size,
                              hipStream_t stream) {
  (void)in_sizes; (void)n_in; (void)out_size; (void)ws_size;
  const float* input   = (const float*)d_in[0];
  const int*   perm    = (const int*)  d_in[1];
  const float* to_kv_w = (const float*)d_in[2];   // Wk = rows 0..767, Wv = rows 768..1535
  const float* to_q_w  = (const float*)d_in[3];
  const float* proj_w  = (const float*)d_in[4];
  const float* proj_b  = (const float*)d_in[5];
  const float* norm_w  = (const float*)d_in[6];
  const float* norm_b  = (const float*)d_in[7];

  float* out  = (float*)d_out;    // [0:98304) out, [98304:159744) attn, [159744:221184) masked
  float* bufA = (float*)d_ws;
  float* bufB = bufA + B_SZ * C_DIM;

  const dim3 ggrid(C_DIM / 16, B_SZ / 16);        // 48 x 8 tiles
  const dim3 gblk(32);

  // T1 = Q_SCALE * q_in @ to_q_w^T   (A = token 0 of each batch, lda = 481*768)
  dv_wmma_gemm<1><<<ggrid, gblk, 0, stream>>>(input, SEQ * C_DIM,
                                              to_q_w, C_DIM,
                                              Q_SCALE, nullptr, bufA, C_DIM);
  // qt = T1 @ Wk
  dv_wmma_gemm<0><<<ggrid, gblk, 0, stream>>>(bufA, C_DIM,
                                              to_kv_w, C_DIM,
                                              1.0f, nullptr, bufB, C_DIM);
  // attn -> softmax/median/mask -> xw
  dv_attn_stream<<<B_SZ, 256, 0, stream>>>(input, perm, bufB,
                                           bufA, out + B_SZ * C_DIM,
                                           out + B_SZ * C_DIM + B_SZ * N_TOK);
  // o1 = xw @ Wv^T
  dv_wmma_gemm<1><<<ggrid, gblk, 0, stream>>>(bufA, C_DIM,
                                              to_kv_w + C_DIM * C_DIM, C_DIM,
                                              1.0f, nullptr, bufB, C_DIM);
  // o2 = o1 @ proj_w^T + proj_b
  dv_wmma_gemm<1><<<ggrid, gblk, 0, stream>>>(bufB, C_DIM,
                                              proj_w, C_DIM,
                                              1.0f, proj_b, bufA, C_DIM);
  // layernorm -> final out
  dv_layernorm<<<B_SZ, 256, 0, stream>>>(bufA, norm_w, norm_b, out);
}
